// GroupedQueryAttention_22625887715428
// MI455X (gfx1250) — matrix operand order repaired and checked by reading, not compiled, not run on hardware
//
#include <hip/hip_runtime.h>

typedef __attribute__((ext_vector_type(16))) _Float16 v16h;
typedef __attribute__((ext_vector_type(8)))  _Float16 v8h;
typedef __attribute__((ext_vector_type(4)))  _Float16 v4h;
typedef __attribute__((ext_vector_type(2)))  _Float16 v2h;
typedef __attribute__((ext_vector_type(8)))  float    v8f;
typedef __attribute__((ext_vector_type(4)))  float    v4f;
typedef __attribute__((ext_vector_type(4)))  unsigned int u32x4;
typedef __attribute__((ext_vector_type(4)))  int      i32x4;
typedef __attribute__((ext_vector_type(8)))  int      i32x8;

constexpr int Bb  = 2;
constexpr int Ss  = 2048;
constexpr int Dd  = 2048;
constexpr int HQ  = 32;
constexpr int HKV = 8;
constexpr int HD  = 64;   // head dim
// NREP = HQ/HKV = 4

__device__ __forceinline__ v8f wmma16x16x32(v16h a, v16h b, v8f c) {
  return __builtin_amdgcn_wmma_f32_16x16x32_f16(
      /*neg_a=*/false, a, /*neg_b=*/false, b,
      /*c_mod=*/(short)0, c, /*reuse_a=*/false, /*reuse_b=*/false);
}

// ---------------------------------------------------------------- TDM helper
// 2D tile DMA: global (row-major, f16) -> LDS, with LDS row padding:
// rows of 32 f16 (16 DWORDs) land at a 40-f16 (80B) stride via
// pad_interval=16 DWORDs (code 3), pad_amount=4 DWORDs (code 3).
__device__ __forceinline__ void tdm_load_tile_f16(
    unsigned int ldsByteOff, const _Float16* gaddr,
    int dim0, int dim1, int stride0, int tile0, int tile1) {
  unsigned long long ga = (unsigned long long)(uintptr_t)gaddr;
  u32x4 g0;
  g0[0] = 1u;                                   // count=1, user mode
  g0[1] = ldsByteOff;                           // lds_addr
  g0[2] = (unsigned int)ga;                     // global_addr[31:0]
  g0[3] = (unsigned int)((ga >> 32) & 0x1FFFFFFu) | (2u << 30);  // addr hi | type=2
  i32x8 g1;
  g1[0] = (1 << 16)      // data_size = 2B
        | (1 << 20)      // pad_enable
        | (3 << 22)      // pad_interval: 16 DWORDs
        | (3 << 25);     // pad_amount: 4 DWORDs
  g1[1] = (int)(((unsigned)dim0 & 0xFFFFu) << 16);                    // dim0 lo
  g1[2] = (int)((((unsigned)dim0 >> 16) & 0xFFFFu) |
                (((unsigned)dim1 & 0xFFFFu) << 16));                  // dim0 hi | dim1 lo
  g1[3] = (int)((((unsigned)dim1 >> 16) & 0xFFFFu) |
                (((unsigned)tile0 & 0xFFFFu) << 16));                 // dim1 hi | tile_dim0
  g1[4] = (int)((unsigned)tile1 & 0xFFFFu);                           // tile_dim1 (tile_dim2=0)
  g1[5] = stride0;                                                    // dim0_stride lo
  g1[6] = 0;
  g1[7] = 0;
  i32x4 z4 = {0, 0, 0, 0};
#if __clang_major__ >= 23
  i32x8 z8 = {0, 0, 0, 0, 0, 0, 0, 0};
  __builtin_amdgcn_tensor_load_to_lds(g0, g1, z4, z4, z8, 0);
#else
  __builtin_amdgcn_tensor_load_to_lds(g0, g1, z4, z4, 0);
#endif
}

// ---------------------------------------------------------------- convert
__global__ void cvt_f32_to_f16(const float* __restrict__ in,
                               _Float16* __restrict__ out, int n4) {
  int i = blockIdx.x * blockDim.x + threadIdx.x;
  int stride = gridDim.x * blockDim.x;
  for (; i < n4; i += stride) {
    v4f x = ((const v4f*)in)[i];
    v4h y;
    y.x = (_Float16)x.x; y.y = (_Float16)x.y;
    y.z = (_Float16)x.z; y.w = (_Float16)x.w;
    ((v4h*)out)[i] = y;
  }
}

// ---------------------------------------------------------------- GEMM
// C[M,N] = A[M,K] * W[N,K]^T (nn.Linear). M,N multiples of 128, K of 32.
// Block: 256 threads = 8 waves. Block tile 128x128, K step 32.
// Staging: double-buffered Tensor Data Mover loads into padded LDS.
// LDS layout (dynamic, base offset 0):
//   buf b at b*20480:  A tile 128x40 f16 (10240B), then B tile 128x40 f16.
template <bool OUT_F16>
__global__ __launch_bounds__(256)
void gemm_nt_wmma(const _Float16* __restrict__ A,
                  const _Float16* __restrict__ W,
                  float* __restrict__ Cf, _Float16* __restrict__ Ch,
                  int M, int N, int K) {
  extern __shared__ _Float16 smem[];   // 2 * 20480 bytes

  const int tid  = threadIdx.x;
  const int wave = tid >> 5, lane = tid & 31;
  const int half = lane >> 4, ln = lane & 15;
  const int wm = wave >> 1, wn = wave & 1;     // 4x2 wave grid
  const int m0 = blockIdx.y * 128, n0 = blockIdx.x * 128;

  v8f acc[2][4];
#pragma unroll
  for (int mi = 0; mi < 2; ++mi)
#pragma unroll
    for (int ni = 0; ni < 4; ++ni)
#pragma unroll
      for (int r = 0; r < 8; ++r) acc[mi][ni][r] = 0.0f;

  const int nk = K >> 5;
  // preload tile 0 into buffer 0 (A tile + B tile)
  if (wave == 0) {
    tdm_load_tile_f16(0u,     A + (size_t)m0 * K, K, M, K, 32, 128);
    tdm_load_tile_f16(10240u, W + (size_t)n0 * K, K, N, K, 32, 128);
  }

  int buf = 0;
  for (int t = 0; t < nk; ++t) {
    if (wave == 0) {
      if (t + 1 < nk) {
        const int kb = (t + 1) << 5;
        const unsigned nb = (unsigned)(buf ^ 1) * 20480u;
        tdm_load_tile_f16(nb,          A + (size_t)m0 * K + kb, K, M, K, 32, 128);
        tdm_load_tile_f16(nb + 10240u, W + (size_t)n0 * K + kb, K, N, K, 32, 128);
        __builtin_amdgcn_s_wait_tensorcnt(2);  // older pair (current tile) done
      } else {
        __builtin_amdgcn_s_wait_tensorcnt(0);
      }
    }
    __syncthreads();

    const _Float16* ldsA = smem + (size_t)buf * 10240;          // elems: 20480B/2
    const _Float16* ldsB = smem + (size_t)buf * 10240 + 5120;   // +10240B

    // A fragments (16x32): lane m=ln; K = 2p + 16*half
    v16h af[2];
#pragma unroll
    for (int mi = 0; mi < 2; ++mi) {
      const _Float16* rp = ldsA + (wm * 32 + mi * 16 + ln) * 40;
#pragma unroll
      for (int p = 0; p < 8; ++p) {
        int k0 = 2 * p + 16 * half;
        v2h pr = *(const v2h*)(rp + k0);
        af[mi][2 * p] = pr.x; af[mi][2 * p + 1] = pr.y;
      }
    }
    // B fragments (32x16, B[k][n] = W[n][k]): lane n=ln; K = 2p + 16*half
    v16h bf[4];
#pragma unroll
    for (int ni = 0; ni < 4; ++ni) {
      const _Float16* rp = ldsB + (wn * 64 + ni * 16 + ln) * 40;
#pragma unroll
      for (int p = 0; p < 8; ++p) {
        int k0 = 2 * p + 16 * half;
        v2h pr = *(const v2h*)(rp + k0);
        bf[ni][2 * p] = pr.x; bf[ni][2 * p + 1] = pr.y;
      }
    }
#pragma unroll
    for (int mi = 0; mi < 2; ++mi)
#pragma unroll
      for (int ni = 0; ni < 4; ++ni)
        acc[mi][ni] = wmma16x16x32(af[mi], bf[ni], acc[mi][ni]);

    __syncthreads();   // everyone done reading buf before DMA overwrites it
    buf ^= 1;
  }

  // C layout: VGPR r -> M = r + 8*half, N = ln
#pragma unroll
  for (int mi = 0; mi < 2; ++mi)
#pragma unroll
    for (int ni = 0; ni < 4; ++ni)
#pragma unroll
      for (int r = 0; r < 8; ++r) {
        int row = m0 + wm * 32 + mi * 16 + r + 8 * half;
        int col = n0 + wn * 64 + ni * 16 + ln;
        if constexpr (OUT_F16)
          Ch[(size_t)row * N + col] = (_Float16)acc[mi][ni][r];
        else
          Cf[(size_t)row * N + col] = acc[mi][ni][r];
      }
}

// ---------------------------------------------------------------- flash attention
// One wave per (b, head, 16-query tile). 32 keys per iteration.
// q/k/v layouts: q [b,s,HQ,HD] f16, k/v [b,s,HKV,HD] f16, o [b,s,HQ,HD] f16.
__global__ __launch_bounds__(256)
void flash_attn_wmma(const _Float16* __restrict__ q,
                     const _Float16* __restrict__ kk,
                     const _Float16* __restrict__ vv,
                     const float* __restrict__ slopes,
                     _Float16* __restrict__ o) {
  __shared__ __align__(16) _Float16 pTile[8][16 * 32];  // per-wave P scratch

  const int tid = threadIdx.x, wave = tid >> 5, lane = tid & 31;
  const int half = lane >> 4, ln = lane & 15;
  const int gw = blockIdx.x * 8 + wave;       // 0..8191
  const int qt = gw & 127;                    // S/16 = 128 query tiles
  const int h  = (gw >> 7) & 31;              // HQ
  const int b  = gw >> 12;                    // B
  const int q0 = qt << 4;
  const int kvh = h >> 2;                     // NREP = 4
  const float slope = slopes[h];
  const float scale = 0.125f;                 // 1/sqrt(64)

  // Q fragments (A-layout), 2 x (16 queries x 32 hd)
  v16h aq[2];
  const _Float16* qrow = q + ((size_t)(b * Ss + q0 + ln) * HQ + h) * HD;
#pragma unroll
  for (int f = 0; f < 2; ++f)
#pragma unroll
    for (int p = 0; p < 8; ++p) {
      int k0 = 32 * f + 2 * p + 16 * half;
      v2h pr = *(const v2h*)(qrow + k0);
      aq[f][2 * p] = pr.x; aq[f][2 * p + 1] = pr.y;
    }

  float mrow[8], lrow[8], fac[8];
  v8f acc[4];
#pragma unroll
  for (int r = 0; r < 8; ++r) { mrow[r] = -1e30f; lrow[r] = 0.0f; }
#pragma unroll
  for (int t = 0; t < 4; ++t)
#pragma unroll
    for (int r = 0; r < 8; ++r) acc[t][r] = 0.0f;

  const size_t kvBase = (size_t)b * Ss;
  const int nIter = (q0 + 16 + 31) >> 5;  // causal: keys 0..q0+15

  for (int it = 0; it < nIter; ++it) {
    const int kStart = it * 32;

    // ---- S = Q K^T for two 16-key sub-tiles
    v8f sfr[2];
#pragma unroll
    for (int s = 0; s < 2; ++s) {
      int krow = kStart + s * 16 + ln;
      if (krow > Ss - 1) krow = Ss - 1;  // OOB-safe; masked below anyway
      const _Float16* kr = kk + ((kvBase + krow) * HKV + kvh) * HD;
      __builtin_prefetch((const void*)(kr + 32 * HKV * HD), 0, 0);  // next K tile
      v16h bk0, bk1;  // hd 0..31 / 32..63, B-layout K = 2p + 16*half
#pragma unroll
      for (int p = 0; p < 8; ++p) {
        int k0 = 2 * p + 16 * half;
        v2h p0 = *(const v2h*)(kr + k0);
        v2h p1 = *(const v2h*)(kr + 32 + k0);
        bk0[2 * p] = p0.x; bk0[2 * p + 1] = p0.y;
        bk1[2 * p] = p1.x; bk1[2 * p + 1] = p1.y;
      }
      v8f c;
#pragma unroll
      for (int r = 0; r < 8; ++r) c[r] = 0.0f;
      c = wmma16x16x32(aq[0], bk0, c);
      c = wmma16x16x32(aq[1], bk1, c);
      sfr[s] = c;
    }

    // ---- online softmax over the 32 new columns
#pragma unroll
    for (int r = 0; r < 8; ++r) {
      int qi  = q0 + r + 8 * half;
      int kj0 = kStart + ln, kj1 = kj0 + 16;
      float x0 = sfr[0][r] * scale + slope * (float)(kj0 - qi);
      float x1 = sfr[1][r] * scale + slope * (float)(kj1 - qi);
      if (kj0 > qi) x0 = -1e30f;
      if (kj1 > qi) x1 = -1e30f;
      float mx = fmaxf(x0, x1);
      mx = fmaxf(mx, __shfl_xor(mx, 1, 32));
      mx = fmaxf(mx, __shfl_xor(mx, 2, 32));
      mx = fmaxf(mx, __shfl_xor(mx, 4, 32));
      mx = fmaxf(mx, __shfl_xor(mx, 8, 32));
      float mnew = fmaxf(mrow[r], mx);
      float f  = __expf(mrow[r] - mnew);
      float p0 = __expf(x0 - mnew);
      float p1 = __expf(x1 - mnew);
      float rs = p0 + p1;
      rs += __shfl_xor(rs, 1, 32);
      rs += __shfl_xor(rs, 2, 32);
      rs += __shfl_xor(rs, 4, 32);
      rs += __shfl_xor(rs, 8, 32);
      lrow[r] = lrow[r] * f + rs;
      mrow[r] = mnew;
      fac[r]  = f;
      // P in row-major LDS (query m x 32 keys), f16
      pTile[wave][(r + 8 * half) * 32 + ln]      = (_Float16)p0;
      pTile[wave][(r + 8 * half) * 32 + 16 + ln] = (_Float16)p1;
    }

    // ---- rescale O accumulators
#pragma unroll
    for (int t = 0; t < 4; ++t)
#pragma unroll
      for (int r = 0; r < 8; ++r) acc[t][r] *= fac[r];

    // ---- reload P as an A fragment (16x32)
    v16h ap;
    {
      const _Float16* pr = &pTile[wave][ln * 32];
#pragma unroll
      for (int p = 0; p < 8; ++p) {
        int k0 = 2 * p + 16 * half;
        v2h pp = *(const v2h*)(pr + k0);
        ap[2 * p] = pp.x; ap[2 * p + 1] = pp.y;
      }
    }

    // ---- O += P * V  (V is [key][hd]: B[k][n] direct, n = hd column)
#pragma unroll
    for (int t = 0; t < 4; ++t) {
      v16h bv;
#pragma unroll
      for (int i = 0; i < 16; ++i) {
        int kd = i + 16 * half;
        int vrow = kStart + kd;
        if (vrow > Ss - 1) vrow = Ss - 1;  // masked keys have P==0
        const _Float16* vp = vv + ((kvBase + vrow) * HKV + kvh) * HD + t * 16 + ln;
        if (i == 0) __builtin_prefetch((const void*)(vp + 32 * HKV * HD), 0, 0);
        bv[i] = *vp;
      }
      acc[t] = wmma16x16x32(ap, bv, acc[t]);
    }
  }

  // ---- epilogue: normalize and store f16 [b,s,h,hd]
#pragma unroll
  for (int r = 0; r < 8; ++r) {
    float inv = 1.0f / lrow[r];
    int qi = q0 + r + 8 * half;
    _Float16* orow = o + ((size_t)(b * Ss + qi) * HQ + h) * HD;
#pragma unroll
    for (int t = 0; t < 4; ++t)
      orow[t * 16 + ln] = (_Float16)(acc[t][r] * inv);
  }
}

// ---------------------------------------------------------------- launch
extern "C" void kernel_launch(void* const* d_in, const int* in_sizes, int n_in,
                              void* d_out, int out_size, void* d_ws, size_t ws_size,
                              hipStream_t stream) {
  (void)in_sizes; (void)n_in; (void)out_size; (void)ws_size;
  const float* x      = (const float*)d_in[0];
  const float* wq     = (const float*)d_in[1];
  const float* wk     = (const float*)d_in[2];
  const float* wv     = (const float*)d_in[3];
  const float* wo     = (const float*)d_in[4];
  const float* slopes = (const float*)d_in[5];
  float* out = (float*)d_out;

  const int M  = Bb * Ss;          // 4096
  const int Nq = HQ * HD;          // 2048
  const int Nk = HKV * HD;         // 512

  char* ws = (char*)d_ws;
  size_t off = 0;
  auto take = [&](size_t elems) {
    _Float16* p = (_Float16*)(ws + off);
    off = (off + elems * sizeof(_Float16) + 255) & ~(size_t)255;
    return p;
  };
  _Float16* xh  = take((size_t)M * Dd);
  _Float16* wqh = take((size_t)Nq * Dd);
  _Float16* wkh = take((size_t)Nk * Dd);
  _Float16* wvh = take((size_t)Nk * Dd);
  _Float16* woh = take((size_t)Dd * Dd);
  _Float16* qh  = take((size_t)M * Nq);
  _Float16* kh  = take((size_t)M * Nk);
  _Float16* vh  = take((size_t)M * Nk);
  _Float16* oh  = take((size_t)M * Nq);

  auto cvt = [&](const float* src, _Float16* dst, size_t n) {
    int n4 = (int)(n / 4);
    int blocks = (n4 + 255) / 256;
    cvt_f32_to_f16<<<blocks, 256, 0, stream>>>(src, dst, n4);
  };
  cvt(x,  xh,  (size_t)M * Dd);
  cvt(wq, wqh, (size_t)Nq * Dd);
  cvt(wk, wkh, (size_t)Nk * Dd);
  cvt(wv, wvh, (size_t)Nk * Dd);
  cvt(wo, woh, (size_t)Dd * Dd);

  const size_t gemmLds = 2u * 20480u;  // double-buffered A+B tiles

  // projections: Q, K, V (f16 outputs)
  dim3 gq(Nq / 128, M / 128);
  gemm_nt_wmma<true><<<gq, 256, gemmLds, stream>>>(xh, wqh, nullptr, qh, M, Nq, Dd);
  dim3 gk(Nk / 128, M / 128);
  gemm_nt_wmma<true><<<gk, 256, gemmLds, stream>>>(xh, wkh, nullptr, kh, M, Nk, Dd);
  gemm_nt_wmma<true><<<gk, 256, gemmLds, stream>>>(xh, wvh, nullptr, vh, M, Nk, Dd);

  // fused flash attention: 8192 wave-tiles, 8 waves/block
  flash_attn_wmma<<<1024, 256, 0, stream>>>(qh, kh, vh, slopes, oh);

  // output projection (f32 -> d_out)
  gemm_nt_wmma<false><<<gq, 256, gemmLds, stream>>>(oh, woh, out, nullptr, M, Dd, Dd);
}
